// ScaledDotProductAttention_1760936591442
// MI455X (gfx1250) — matrix operand order repaired and checked by reading, not compiled, not run on hardware
//
#include <hip/hip_runtime.h>
#include <math.h>

// ---------------------------------------------------------------------------
// Scaled dot-product attention for MI455X (gfx1250, wave32).
//   B=4, H=16, S=2048, D=64. Outputs: context [B,H,S,64] ++ attn [B,H,S,S].
// One wave handles one (head, 16-row Q tile). bf16 WMMA (v_wmma_f32_16x16x32_bf16)
// with f32 accumulation; two-pass softmax: pass A keeps LANE-LOCAL online
// (max,sum) stats (no cross-lane traffic in the hot loop, one merge at the
// end); pass B recomputes QK^T, writes final attn probabilities, and
// accumulates P@V via WMMA with P staged through a tiny per-wave LDS tile.
// 1/sqrt(d) is folded (exactly, power of two) into the Q fragment; 1/l is
// folded into a log-sum-exp offset m' = m + ln(l), so the pass-B inner loop
// is a single v_exp per element.
// ---------------------------------------------------------------------------

typedef __attribute__((ext_vector_type(16))) __bf16       v16bf;
typedef __attribute__((ext_vector_type(8)))  float        v8f;
typedef __attribute__((ext_vector_type(4)))  unsigned int v4u;
typedef __attribute__((ext_vector_type(4)))  float        v4f;

#define S_LEN   2048
#define D_HEAD  64
#define NTILE   (S_LEN / 16)     // 128 k-tiles of 16 columns
#define BH      64               // B*H
#define QTILES  (S_LEN / 16)     // 128 q-tiles per head
#define LDS_STRIDE 40            // ushorts per LDS row: 80B (16B aligned, bank spread)

// Native f32 -> bf16 (lowers to packed v_cvt_pk_bf16_f32)
static __device__ __forceinline__ __bf16 f2bf(float f) { return (__bf16)f; }
static __device__ __forceinline__ unsigned short f2bf_bits(float f) {
  return __builtin_bit_cast(unsigned short, (__bf16)f);
}

union BF16x16 {
  v16bf  v;
  __bf16 b[16];
  v4u    q[2];
};

// 16 f32 from p[0..7] and p[16..23] -> bf16 B-fragment (vectorized 128-bit global loads;
// adjacent pairs convert via packed bf16 cvt)
static __device__ __forceinline__ v16bf cvt16(const float* __restrict__ p) {
  BF16x16 r;
  const v4f* p4 = (const v4f*)p;
#pragma unroll
  for (int i = 0; i < 4; ++i) {
    v4f x = p4[i < 2 ? i : i + 2];
    r.b[4 * i + 0] = f2bf(x.x);
    r.b[4 * i + 1] = f2bf(x.y);
    r.b[4 * i + 2] = f2bf(x.z);
    r.b[4 * i + 3] = f2bf(x.w);
  }
  return r.v;
}

// A-fragment (16x32 bf16) per-lane layout: lane<16 row M=lane, K locals
// {0..7,16..23}; lane>=16 row M=lane-16, K locals {8..15,24..31}. Two
// contiguous 8-float runs per lane. `scale` (0.125 = 1/sqrt(64), a power of
// two) is applied exactly in f32 before the bf16 convert.
static __device__ __forceinline__ v16bf load_a_f32(const float* __restrict__ row,
                                                   int d0, int hw, float scale) {
  BF16x16 r;
  const float* p1 = row + d0 + 8 * hw;
  const float* p2 = p1 + 16;
#pragma unroll
  for (int i = 0; i < 8; ++i) {
    r.b[i]     = f2bf(p1[i] * scale);
    r.b[8 + i] = f2bf(p2[i] * scale);
  }
  return r.v;
}

// V B-fragment (32x16): lane<16 holds column N=lane for K locals 0..15,
// lane>=16 holds N=lane-16 for K locals 16..31 (column gather, half-wave coalesced).
static __device__ __forceinline__ v16bf load_v_frag(const float* __restrict__ Vh,
                                                    int kk0, int n, int nl, int hw) {
  BF16x16 r;
  const float* base = Vh + (size_t)(kk0 + 16 * hw) * D_HEAD + n * 16 + nl;
#pragma unroll
  for (int e = 0; e < 16; ++e) r.b[e] = f2bf(base[(size_t)e * D_HEAD]);
  return r.v;
}

static __device__ __forceinline__ v8f wmma_bf16(v16bf a, v16bf b, v8f c) {
  return __builtin_amdgcn_wmma_f32_16x16x32_bf16(
      /*neg_a=*/false, a, /*neg_b=*/false, b,
      /*c_mod=*/(short)0, c, /*reuse_a=*/false, /*reuse_b=*/false);
}

// One 16x16 masked score tile (Q pre-scaled): element r -> (M=r+8*hw, N=nl).
static __device__ __forceinline__ v8f score_tile(const float* __restrict__ Kh,
                                                 const unsigned char* __restrict__ Mh,
                                                 v16bf a_lo, v16bf a_hi,
                                                 int k0, int nl, int hw) {
  const float* krow = Kh + (size_t)(k0 + nl) * D_HEAD + 8 * hw;
  v16bf b_lo = cvt16(krow);        // d = 8*hw .. +7, 16 + 8*hw .. +7
  v16bf b_hi = cvt16(krow + 32);   // d = 32 + 8*hw .. +7, 48 + 8*hw .. +7
  v8f c = {};
  c = wmma_bf16(a_lo, b_lo, c);
  c = wmma_bf16(a_hi, b_hi, c);
#pragma unroll
  for (int r = 0; r < 8; ++r) {
    int M = r + 8 * hw;
    unsigned char mk = Mh[(size_t)M * S_LEN + k0 + nl];   // mask==true -> -1e9
    c[r] = mk ? -1.0e9f : c[r];
  }
  return c;
}

__global__ __launch_bounds__(128) void
sdpa_wmma_kernel(const float* __restrict__ Q, const float* __restrict__ K,
                 const float* __restrict__ V, const unsigned char* __restrict__ mask,
                 float* __restrict__ ctx_out, float* __restrict__ attn_out) {
  __shared__ __align__(16) unsigned short ldsP[4 * 16 * LDS_STRIDE];

  const int tid  = threadIdx.x;
  const int wave = tid >> 5;
  const int lane = tid & 31;
  const int hw   = lane >> 4;    // half-wave
  const int nl   = lane & 15;    // N / column lane index

  const int tile = blockIdx.x * 4 + wave;     // 0..8191
  const int bh   = tile >> 7;                 // head index 0..63
  const int q0   = (tile & (QTILES - 1)) * 16;

  const float* Qh = Q + ((size_t)bh * S_LEN + q0) * D_HEAD;
  const float* Kh = K + (size_t)bh * S_LEN * D_HEAD;
  const float* Vh = V + (size_t)bh * S_LEN * D_HEAD;
  const unsigned char* Mh = mask + (size_t)bh * S_LEN * S_LEN + (size_t)q0 * S_LEN;
  float* Ah = attn_out + (size_t)bh * S_LEN * S_LEN + (size_t)q0 * S_LEN;
  float* Ch = ctx_out + ((size_t)bh * S_LEN + q0) * D_HEAD;

  // Q tile resident as two A-fragments (K = 0..31, 32..63), pre-scaled by 1/8.
  const float* qrow = Qh + nl * D_HEAD;
  const v16bf a_lo = load_a_f32(qrow, 0, hw, 0.125f);
  const v16bf a_hi = load_a_f32(qrow, 32, hw, 0.125f);

  // -------- Pass A: lane-local online softmax stats over this lane's columns
  float m[8], l[8];
#pragma unroll
  for (int r = 0; r < 8; ++r) { m[r] = -3.0e38f; l[r] = 0.0f; }

  for (int kt = 0; kt < NTILE; ++kt) {
    const int k0 = kt * 16;
    if (kt + 1 < NTILE) {   // stream-ahead hints -> global_prefetch_b8
      __builtin_prefetch(Kh + (size_t)(k0 + 16 + nl) * D_HEAD, 0, 1);
      __builtin_prefetch(Mh + (size_t)(8 * hw) * S_LEN + k0 + 16 + nl, 0, 1);
    }
    v8f s = score_tile(Kh, Mh, a_lo, a_hi, k0, nl, hw);
#pragma unroll
    for (int r = 0; r < 8; ++r) {
      float mn = fmaxf(m[r], s[r]);
      l[r] = l[r] * __expf(m[r] - mn) + __expf(s[r] - mn);   // TRANS co-exec w/ WMMA
      m[r] = mn;
    }
  }
  // Single cross-lane merge of (m,l) across the 16 lanes of each half-wave.
#pragma unroll
  for (int step = 8; step >= 1; step >>= 1) {
#pragma unroll
    for (int r = 0; r < 8; ++r) {
      float om = __shfl_xor(m[r], step, 32);
      float ol = __shfl_xor(l[r], step, 32);
      float mn = fmaxf(m[r], om);
      l[r] = l[r] * __expf(m[r] - mn) + ol * __expf(om - mn);
      m[r] = mn;
    }
  }
  // log-sum-exp offset: p = exp(s - m - ln(l)) — one v_exp per element in pass B
  float mp[8];
#pragma unroll
  for (int r = 0; r < 8; ++r) mp[r] = m[r] + __logf(l[r]);

  // -------- Pass B: recompute scores, emit attn, accumulate P @ V --------
  v8f acc0 = {}, acc1 = {}, acc2 = {}, acc3 = {};
  unsigned short* myP = ldsP + wave * (16 * LDS_STRIDE);

  for (int kp = 0; kp < NTILE / 2; ++kp) {      // 32 k-columns per iteration
#pragma unroll
    for (int t = 0; t < 2; ++t) {
      const int k0 = (2 * kp + t) * 16;
      v8f s = score_tile(Kh, Mh, a_lo, a_hi, k0, nl, hw);
#pragma unroll
      for (int r = 0; r < 8; ++r) {
        const int M = r + 8 * hw;
        float p = __expf(s[r] - mp[r]);
        Ah[(size_t)M * S_LEN + k0 + nl] = p;                 // final attn out
        myP[M * LDS_STRIDE + t * 16 + nl] = f2bf_bits(p);    // stage for A-frag
      }
    }
    // Reload P (16x32 bf16) from LDS in A-fragment layout (DS ops are
    // in-order within a wave; compiler inserts s_wait_dscnt for the RAW).
    BF16x16 pa;
    pa.q[0] = *(const v4u*)(myP + nl * LDS_STRIDE + 16 * hw);
    pa.q[1] = *(const v4u*)(myP + nl * LDS_STRIDE + 8 + 16 * hw);

    const int kk0 = kp * 32;
    acc0 = wmma_bf16(pa.v, load_v_frag(Vh, kk0, 0, nl, hw), acc0);
    acc1 = wmma_bf16(pa.v, load_v_frag(Vh, kk0, 1, nl, hw), acc1);
    acc2 = wmma_bf16(pa.v, load_v_frag(Vh, kk0, 2, nl, hw), acc2);
    acc3 = wmma_bf16(pa.v, load_v_frag(Vh, kk0, 3, nl, hw), acc3);
  }

  // -------- store context: element r -> (M=r+8*hw, d=n*16+nl) --------
#pragma unroll
  for (int r = 0; r < 8; ++r) {
    const int M = r + 8 * hw;
    float* crow = Ch + (size_t)M * D_HEAD + nl;
    crow[0]  = acc0[r];
    crow[16] = acc1[r];
    crow[32] = acc2[r];
    crow[48] = acc3[r];
  }
}

extern "C" void kernel_launch(void* const* d_in, const int* in_sizes, int n_in,
                              void* d_out, int out_size, void* d_ws, size_t ws_size,
                              hipStream_t stream) {
  (void)in_sizes; (void)n_in; (void)out_size; (void)d_ws; (void)ws_size;
  const float*         Q    = (const float*)d_in[0];
  const float*         K    = (const float*)d_in[1];
  const float*         V    = (const float*)d_in[2];
  const unsigned char* mask = (const unsigned char*)d_in[3];  // jnp.bool_ = 1 byte

  float* out  = (float*)d_out;
  float* ctx  = out;                                       // [B,H,S,64]
  float* attn = out + (size_t)BH * S_LEN * D_HEAD;         // [B,H,S,S]

  const int total_waves  = BH * QTILES;                    // 8192
  const int blocks       = total_waves / 4;                // 4 waves / block
  sdpa_wmma_kernel<<<dim3(blocks), dim3(128), 0, stream>>>(Q, K, V, mask, ctx, attn);
}
